// BertFlashFWSVDBlock_2250562863188
// MI455X (gfx1250) — matrix operand order repaired and checked by reading, not compiled, not run on hardware
//
#include <hip/hip_runtime.h>

typedef __attribute__((ext_vector_type(16))) __bf16 v16bf;
typedef __attribute__((ext_vector_type(8)))  __bf16 bf16x8;
typedef __attribute__((ext_vector_type(8)))  float  v8f;

#define DEV __device__ __forceinline__

DEV __bf16 f2bf(float f) {
  unsigned u = __float_as_uint(f);
  u += 0x7FFFu + ((u >> 16) & 1u);  // round-to-nearest-even
  unsigned short hs = (unsigned short)(u >> 16);
  __bf16 r;
  __builtin_memcpy(&r, &hs, 2);
  return r;
}

DEV v16bf concat8(bf16x8 lo, bf16x8 hi) {
  return __builtin_shufflevector(lo, hi, 0, 1, 2, 3, 4, 5, 6, 7,
                                 8, 9, 10, 11, 12, 13, 14, 15);
}

// A-fragment of a 16x32 bf16 tile (MxK), row-major source, leading dim `ld`.
// lane<16 -> row=lane, K 0..15; lane>=16 -> same row, K 16..31.
DEV v16bf load_afrag(const __bf16* p, int ld, int lane) {
  const int row = lane & 15;
  const int kb  = (lane >> 4) * 16;
  const __bf16* q = p + row * ld;
  bf16x8 lo = *(const bf16x8*)(q + kb);
  bf16x8 hi = *(const bf16x8*)(q + kb + 8);
  return concat8(lo, hi);
}

// B-fragment of a 32x16 (KxN) tile gathered from an N-major buffer:
// element (k,n) at p[n*ld + k]. Lane<16: n=lane, K=0..15; lane>=16: n=lane-16,
// K=16..31. 32 contiguous bytes per lane.
DEV v16bf load_bfrag(const __bf16* p, int ld, int lane) {
  const int n  = lane & 15;
  const int kb = (lane >> 4) * 16;
  const __bf16* q = p + n * ld + kb;
  bf16x8 lo = *(const bf16x8*)(q);
  bf16x8 hi = *(const bf16x8*)(q + 8);
  return concat8(lo, hi);
}

DEV v8f wmma_bf16(v16bf a, v16bf b, v8f c) {
  return __builtin_amdgcn_wmma_f32_16x16x32_bf16(false, a, false, b,
                                                 (short)0, c, false, false);
}

// ---------------------------------------------------------------------------
// elementwise fp32 -> bf16
__global__ void f2b_kernel(const float* __restrict__ in,
                           __bf16* __restrict__ out, long long n) {
  long long i = (long long)blockIdx.x * blockDim.x + threadIdx.x;
  const long long stride = (long long)gridDim.x * blockDim.x;
  for (; i < n; i += stride) out[i] = f2bf(in[i]);
}

// P (H, D, R) fp32 -> N-major (H*R, D) bf16: out[(h*R+r)*D + d] = in[(h*D+d)*R + r]
__global__ void repackP_kernel(const float* __restrict__ in,
                               __bf16* __restrict__ out,
                               int H, int D, int R) {
  const long long n = (long long)H * D * R;
  long long i = (long long)blockIdx.x * blockDim.x + threadIdx.x;
  const long long stride = (long long)gridDim.x * blockDim.x;
  for (; i < n; i += stride) {
    const int r = (int)(i % R);
    const long long t = i / R;
    const int d = (int)(t % D);
    const int h = (int)(t / D);
    out[(h * R + r) * D + d] = f2bf(in[i]);
  }
}

// V (H, R, DH) fp32 -> per-head N-major (H, DH, R) bf16
__global__ void repackV_kernel(const float* __restrict__ in,
                               __bf16* __restrict__ out,
                               int H, int R, int DH) {
  const long long n = (long long)H * R * DH;
  long long i = (long long)blockIdx.x * blockDim.x + threadIdx.x;
  const long long stride = (long long)gridDim.x * blockDim.x;
  for (; i < n; i += stride) {
    const int dh = (int)(i % DH);
    const long long t = i / DH;
    const int r = (int)(t % R);
    const int h = (int)(t / R);
    out[(h * DH + dh) * R + r] = f2bf(in[i]);
  }
}

// weight (Rw, Cl) fp32 row-major -> (Cl, Rw) bf16 (N-major for GEMM B operand)
__global__ void transposeW_kernel(const float* __restrict__ in,
                                  __bf16* __restrict__ out, int Rw, int Cl) {
  const long long n = (long long)Rw * Cl;
  long long i = (long long)blockIdx.x * blockDim.x + threadIdx.x;
  const long long stride = (long long)gridDim.x * blockDim.x;
  for (; i < n; i += stride) {
    const int c = (int)(i % Cl);
    const int r = (int)(i / Cl);
    out[(long long)c * Rw + r] = f2bf(in[i]);
  }
}

// (HB, Mseq, DH) -> (HB, DH, Mseq) bf16 transpose (for attention V)
__global__ void transpose_kernel(const __bf16* __restrict__ in,
                                 __bf16* __restrict__ out,
                                 int Mseq, int DH, long long n) {
  long long i = (long long)blockIdx.x * blockDim.x + threadIdx.x;
  const long long stride = (long long)gridDim.x * blockDim.x;
  for (; i < n; i += stride) {
    const int dh = (int)(i % DH);
    const long long t = i / DH;
    const int m = (int)(t % Mseq);
    const long long hb = t / Mseq;
    out[hb * (long long)Mseq * DH + dh * Mseq + m] = in[i];
  }
}

// ---------------------------------------------------------------------------
// Batched GEMM: C[bz] = A[bz] (MxK, row-major) * Bt[bz] (N-major KxN weights)
// 128 threads = 4 waves; block tile 128x64; wave tile 32x64; K-step 32.
// A staged through double-buffered LDS (coalesced HBM); B fragments loaded
// straight from global (weights are L2/WGP$-resident on MI455X's 192MB L2).
__global__ __launch_bounds__(128) void gemm_kernel(
    const __bf16* __restrict__ A, const __bf16* __restrict__ Bt,
    float* __restrict__ Cf, __bf16* __restrict__ Cb,
    const float* __restrict__ bias,
    int N, int K, int lda, int ldc,
    long long sAb, long long sBb, long long sCb,
    int biasBatchStride, int act) {
  __shared__ __bf16 sA[2][128 * 32];
  const int tid = threadIdx.x, lane = tid & 31, wv = tid >> 5;
  const int col0 = blockIdx.x * 64, row0 = blockIdx.y * 128;
  const __bf16* Ap  = A + (long long)blockIdx.z * sAb;
  const __bf16* Btp = Bt + (long long)blockIdx.z * sBb;

  // A-tile copy: 128x32 halves; 4 x 16-byte chunks per thread, straight-line
  const int cr = tid >> 2;         // 0..31
  const int cc = (tid & 3) << 3;   // 0,8,16,24
  const __bf16* ag = Ap + (row0 + cr) * lda + cc;
  const int l0 = cr * 32 + cc;

  const int bn = lane & 15, bkb = (lane >> 4) * 16;
  v8f acc[2][4] = {};

  auto compute = [&](int buf, int kk) {
    const v16bf afL = load_afrag(&sA[buf][(wv * 32) * 32], 32, lane);
    const v16bf afH = load_afrag(&sA[buf][(wv * 32 + 16) * 32], 32, lane);
#pragma unroll
    for (int nt = 0; nt < 4; ++nt) {
      const __bf16* q = Btp + (col0 + nt * 16 + bn) * K + kk + bkb;
      const v16bf bf = concat8(*(const bf16x8*)q, *(const bf16x8*)(q + 8));
      acc[0][nt] = wmma_bf16(afL, bf, acc[0][nt]);
      acc[1][nt] = wmma_bf16(afH, bf, acc[1][nt]);
    }
  };

  // preload tile 0
  bf16x8 t0 = *(const bf16x8*)(ag);
  bf16x8 t1 = *(const bf16x8*)(ag + 32 * lda);
  bf16x8 t2 = *(const bf16x8*)(ag + 64 * lda);
  bf16x8 t3 = *(const bf16x8*)(ag + 96 * lda);
  *(bf16x8*)&sA[0][l0]           = t0;
  *(bf16x8*)&sA[0][l0 + 32 * 32] = t1;
  *(bf16x8*)&sA[0][l0 + 64 * 32] = t2;
  *(bf16x8*)&sA[0][l0 + 96 * 32] = t3;
  __syncthreads();

  int cur = 0;
  int k0 = 0;
  for (; k0 + 32 < K; k0 += 32) {
    const __bf16* an = ag + k0 + 32;      // prefetch next tile
    t0 = *(const bf16x8*)(an);
    t1 = *(const bf16x8*)(an + 32 * lda);
    t2 = *(const bf16x8*)(an + 64 * lda);
    t3 = *(const bf16x8*)(an + 96 * lda);
    compute(cur, k0);                      // wmma on current tile
    const int nxt = cur ^ 1;               // stage next tile
    *(bf16x8*)&sA[nxt][l0]           = t0;
    *(bf16x8*)&sA[nxt][l0 + 32 * 32] = t1;
    *(bf16x8*)&sA[nxt][l0 + 64 * 32] = t2;
    *(bf16x8*)&sA[nxt][l0 + 96 * 32] = t3;
    __syncthreads();
    cur = nxt;
  }
  compute(cur, k0);                        // last tile

  const int hi = lane >> 4, ln = lane & 15;
#pragma unroll
  for (int ms = 0; ms < 2; ++ms)
#pragma unroll
    for (int nt = 0; nt < 4; ++nt) {
      const int col = col0 + nt * 16 + ln;
      const float bvv = bias ? bias[biasBatchStride * blockIdx.z + col] : 0.f;
#pragma unroll
      for (int r = 0; r < 8; ++r) {
        const int rowm = row0 + wv * 32 + ms * 16 + hi * 8 + r;
        float v = acc[ms][nt][r] + bvv;
        if (act == 1) v = 0.5f * v * (1.f + erff(v * 0.70710678118654752f));
        const long long idx = (long long)blockIdx.z * sCb + rowm * ldc + col;
        if (Cf) Cf[idx] = v;
        if (Cb) Cb[idx] = f2bf(v);
      }
    }
}

// ---------------------------------------------------------------------------
// Flash attention: grid (M/64, B, H), 128 threads = 4 waves, 16 rows/wave.
// Q,K in (H,B,M,64) bf16; Vt in (H,B,64,M) bf16; out attn (B*M, 768) bf16.
__global__ __launch_bounds__(128) void attn_kernel(
    const __bf16* __restrict__ Q, const __bf16* __restrict__ Kmat,
    const __bf16* __restrict__ Vt, const float* __restrict__ mask,
    __bf16* __restrict__ attn, int Mseq, int Dmodel, float scale) {
  __shared__ __bf16 sP[4][16 * 64];  // per-wave P tile (C-layout -> A-layout)
  const int mt = blockIdx.x, b = blockIdx.y, h = blockIdx.z;
  const int Bb = gridDim.y;
  const int tid = threadIdx.x, lane = tid & 31, wv = tid >> 5;
  const int hi = lane >> 4, ln = lane & 15;
  const int hb = h * Bb + b;
  const __bf16* Qp = Q + hb * Mseq * 64;
  const __bf16* Kp = Kmat + hb * Mseq * 64;
  const __bf16* Vp = Vt + hb * Mseq * 64;  // (64, Mseq)
  const int row0 = mt * 64 + wv * 16;

  const v16bf qf0 = load_afrag(Qp + row0 * 64, 64, lane);
  const v16bf qf1 = load_afrag(Qp + row0 * 64 + 32, 64, lane);

  float mrow[8], lrow[8];
  v8f o[4] = {};
#pragma unroll
  for (int r = 0; r < 8; ++r) { mrow[r] = -3.0e38f; lrow[r] = 0.f; }

  for (int j0 = 0; j0 < Mseq; j0 += 64) {
    v8f s[4];
#pragma unroll
    for (int nt = 0; nt < 4; ++nt) {
      const int nb = j0 + nt * 16;
      const v16bf kf0 = load_bfrag(Kp + nb * 64, 64, lane);
      const v16bf kf1 = load_bfrag(Kp + nb * 64 + 32, 64, lane);
      v8f t = {};
      t = wmma_bf16(qf0, kf0, t);
      t = wmma_bf16(qf1, kf1, t);
      const float mv = mask[b * Mseq + nb + ln];
#pragma unroll
      for (int r = 0; r < 8; ++r) s[nt][r] = t[r] * scale + mv;
    }
    float rm[8];
#pragma unroll
    for (int r = 0; r < 8; ++r)
      rm[r] = fmaxf(fmaxf(s[0][r], s[1][r]), fmaxf(s[2][r], s[3][r]));
#pragma unroll
    for (int off = 1; off < 16; off <<= 1)
#pragma unroll
      for (int r = 0; r < 8; ++r)
        rm[r] = fmaxf(rm[r], __shfl_xor(rm[r], off, 32));
    float alpha[8], rs[8];
#pragma unroll
    for (int r = 0; r < 8; ++r) {
      const float mn = fmaxf(mrow[r], rm[r]);
      alpha[r] = __expf(mrow[r] - mn);
      mrow[r] = mn;
      rs[r] = 0.f;
    }
#pragma unroll
    for (int nt = 0; nt < 4; ++nt)
#pragma unroll
      for (int r = 0; r < 8; ++r) {
        const float p = __expf(s[nt][r] - mrow[r]);
        s[nt][r] = p;
        rs[r] += p;
      }
#pragma unroll
    for (int off = 1; off < 16; off <<= 1)
#pragma unroll
      for (int r = 0; r < 8; ++r) rs[r] += __shfl_xor(rs[r], off, 32);
#pragma unroll
    for (int r = 0; r < 8; ++r) lrow[r] = lrow[r] * alpha[r] + rs[r];
#pragma unroll
    for (int nt = 0; nt < 4; ++nt)
#pragma unroll
      for (int r = 0; r < 8; ++r) o[nt][r] *= alpha[r];

    __bf16* Pw = sP[wv];
#pragma unroll
    for (int nt = 0; nt < 4; ++nt) {
      const int col = nt * 16 + ln;
#pragma unroll
      for (int r = 0; r < 8; ++r)
        Pw[(hi * 8 + r) * 64 + col] = f2bf(s[nt][r]);
    }
    const v16bf pf0 = load_afrag(Pw, 64, lane);
    const v16bf pf1 = load_afrag(Pw + 32, 64, lane);
#pragma unroll
    for (int nt = 0; nt < 4; ++nt) {
      const __bf16* vb = Vp + (nt * 16) * Mseq + j0;
      const v16bf vf0 = load_bfrag(vb, Mseq, lane);
      const v16bf vf1 = load_bfrag(vb + 32, Mseq, lane);
      o[nt] = wmma_bf16(pf0, vf0, o[nt]);
      o[nt] = wmma_bf16(pf1, vf1, o[nt]);
    }
  }
#pragma unroll
  for (int nt = 0; nt < 4; ++nt)
#pragma unroll
    for (int r = 0; r < 8; ++r) {
      const int m = row0 + hi * 8 + r;
      attn[(b * Mseq + m) * Dmodel + h * 64 + nt * 16 + ln] =
          f2bf(o[nt][r] / lrow[r]);
    }
}

// ---------------------------------------------------------------------------
// Fused residual + LayerNorm: out = g*(z-mu)*rsqrt(var+eps)+beta, z = a + b
__global__ __launch_bounds__(256) void ln_kernel(
    const float* __restrict__ a, const float* __restrict__ bsum,
    const float* __restrict__ g, const float* __restrict__ beta,
    float* __restrict__ outf, __bf16* __restrict__ outb, int D) {
  const int row = blockIdx.x;
  const float* pa = a + (long long)row * D;
  const float* pb = bsum + (long long)row * D;
  float s = 0.f, s2 = 0.f;
  for (int i = threadIdx.x; i < D; i += blockDim.x) {
    const float z = pa[i] + pb[i];
    s += z;
    s2 += z * z;
  }
#pragma unroll
  for (int off = 16; off; off >>= 1) {
    s += __shfl_xor(s, off, 32);
    s2 += __shfl_xor(s2, off, 32);
  }
  __shared__ float rs[8], rs2[8];
  const int wv = threadIdx.x >> 5;
  if ((threadIdx.x & 31) == 0) { rs[wv] = s; rs2[wv] = s2; }
  __syncthreads();
  s = 0.f; s2 = 0.f;
#pragma unroll
  for (int i = 0; i < 8; ++i) { s += rs[i]; s2 += rs2[i]; }
  const float mu = s / D;
  const float var = s2 / D - mu * mu;
  const float rinv = rsqrtf(var + 1e-12f);
  for (int i = threadIdx.x; i < D; i += blockDim.x) {
    const float z = pa[i] + pb[i];
    const float o = g[i] * (z - mu) * rinv + beta[i];
    if (outf) outf[(long long)row * D + i] = o;
    if (outb) outb[(long long)row * D + i] = f2bf(o);
  }
}

// ---------------------------------------------------------------------------
extern "C" void kernel_launch(void* const* d_in, const int* in_sizes, int n_in,
                              void* d_out, int out_size, void* d_ws,
                              size_t ws_size, hipStream_t stream) {
  (void)in_sizes; (void)n_in; (void)out_size; (void)ws_size;
  const int Bb = 8, Mm = 1024, Dd = 768, Hh = 12, DH = 64, Rr = 32;
  const int RF = 384, RW = 384, DFF = 3072;
  const long long BM = (long long)Bb * Mm;  // 8192

  const float* x    = (const float*)d_in[0];
  const float* mask = (const float*)d_in[1];
  const float* Pq   = (const float*)d_in[2];
  const float* Vq   = (const float*)d_in[3];
  const float* bq   = (const float*)d_in[4];
  const float* Pk   = (const float*)d_in[5];
  const float* Vk   = (const float*)d_in[6];
  const float* bk   = (const float*)d_in[7];
  const float* Pv   = (const float*)d_in[8];
  const float* Vv   = (const float*)d_in[9];
  const float* bv   = (const float*)d_in[10];
  const float* Uo   = (const float*)d_in[11];
  const float* Vo   = (const float*)d_in[12];
  const float* bo   = (const float*)d_in[13];
  const float* U1   = (const float*)d_in[14];
  const float* V1   = (const float*)d_in[15];
  const float* b1   = (const float*)d_in[16];
  const float* U2   = (const float*)d_in[17];
  const float* V2   = (const float*)d_in[18];
  const float* b2   = (const float*)d_in[19];
  const float* g1   = (const float*)d_in[20];
  const float* be1  = (const float*)d_in[21];
  const float* g2   = (const float*)d_in[22];
  const float* be2  = (const float*)d_in[23];

  char* ws = (char*)d_ws;
  size_t off = 0;
  auto alloc = [&](size_t bytes) -> void* {
    off = (off + 255) & ~(size_t)255;
    void* p = ws + off;
    off += bytes;
    return p;
  };

  __bf16* xb   = (__bf16*)alloc((size_t)BM * Dd * 2);
  __bf16* Pqt  = (__bf16*)alloc((size_t)Hh * Rr * Dd * 2);  // (H*R, D) N-major
  __bf16* Pkt  = (__bf16*)alloc((size_t)Hh * Rr * Dd * 2);
  __bf16* Pvt  = (__bf16*)alloc((size_t)Hh * Rr * Dd * 2);
  __bf16* Vqt  = (__bf16*)alloc((size_t)Hh * DH * Rr * 2);  // (H, DH, R)
  __bf16* Vkt  = (__bf16*)alloc((size_t)Hh * DH * Rr * 2);
  __bf16* Vvt  = (__bf16*)alloc((size_t)Hh * DH * Rr * 2);
  __bf16* Uot  = (__bf16*)alloc((size_t)RW * Dd * 2);       // (RW, D)
  __bf16* Vot  = (__bf16*)alloc((size_t)Dd * RW * 2);       // (D, RW)
  __bf16* U1t  = (__bf16*)alloc((size_t)RF * Dd * 2);       // (RF, D)
  __bf16* V1t  = (__bf16*)alloc((size_t)DFF * RF * 2);      // (DFF, RF)
  __bf16* U2t  = (__bf16*)alloc((size_t)RF * DFF * 2);      // (RF, DFF)
  __bf16* V2t  = (__bf16*)alloc((size_t)Dd * RF * 2);       // (D, RF)
  __bf16* xPq  = (__bf16*)alloc((size_t)BM * Hh * Rr * 2);
  __bf16* xPk  = (__bf16*)alloc((size_t)BM * Hh * Rr * 2);
  __bf16* xPv  = (__bf16*)alloc((size_t)BM * Hh * Rr * 2);
  const size_t qkvBytes = (size_t)Hh * BM * DH * 2;  // 12.58 MB, 256-aligned
  __bf16* Qb  = (__bf16*)alloc(qkvBytes);
  __bf16* Kb  = (__bf16*)alloc(qkvBytes);
  __bf16* Vb  = (__bf16*)alloc(qkvBytes);
  __bf16* Vtb = (__bf16*)alloc(qkvBytes);
  __bf16* attn  = (__bf16*)alloc((size_t)BM * Dd * 2);
  __bf16* attnU = (__bf16*)alloc((size_t)BM * RW * 2);
  float*  y0    = (float*)alloc((size_t)BM * Dd * 4);
  float*  x1f   = (float*)alloc((size_t)BM * Dd * 4);
  __bf16* x1b   = (__bf16*)alloc((size_t)BM * Dd * 2);
  __bf16* mid   = (__bf16*)alloc((size_t)BM * RF * 2);
  __bf16* ffn2  = (__bf16*)alloc((size_t)BM * RF * 2);
  __bf16* ffn1  = Qb;  // reuse Q/K/V/Vt block (4*12.58MB == BM*DFF*2), dead after attn
  float*  y2    = y0;  // reuse: y0 consumed by LN1 before y2 is produced

  auto gemm = [&](const __bf16* A, const __bf16* Bt, float* Cf, __bf16* Cb,
                  const float* bias, int M, int N, int K, int lda, int ldc,
                  long long sAb, long long sBb, long long sCb,
                  int bStride, int act, int batch) {
    dim3 g(N / 64, M / 128, batch);
    gemm_kernel<<<g, 128, 0, stream>>>(A, Bt, Cf, Cb, bias, N, K, lda, ldc,
                                       sAb, sBb, sCb, bStride, act);
  };

  // --- precision prep: bf16 conversion + N-major weight transposes ---
  f2b_kernel<<<512, 256, 0, stream>>>(x, xb, BM * Dd);
  repackP_kernel<<<288, 256, 0, stream>>>(Pq, Pqt, Hh, Dd, Rr);
  repackP_kernel<<<288, 256, 0, stream>>>(Pk, Pkt, Hh, Dd, Rr);
  repackP_kernel<<<288, 256, 0, stream>>>(Pv, Pvt, Hh, Dd, Rr);
  repackV_kernel<<<96, 256, 0, stream>>>(Vq, Vqt, Hh, Rr, DH);
  repackV_kernel<<<96, 256, 0, stream>>>(Vk, Vkt, Hh, Rr, DH);
  repackV_kernel<<<96, 256, 0, stream>>>(Vv, Vvt, Hh, Rr, DH);
  transposeW_kernel<<<288, 256, 0, stream>>>(Uo, Uot, Dd, RW);
  transposeW_kernel<<<288, 256, 0, stream>>>(Vo, Vot, RW, Dd);
  transposeW_kernel<<<288, 256, 0, stream>>>(U1, U1t, Dd, RF);
  transposeW_kernel<<<512, 256, 0, stream>>>(V1, V1t, RF, DFF);
  transposeW_kernel<<<512, 256, 0, stream>>>(U2, U2t, DFF, RF);
  transposeW_kernel<<<288, 256, 0, stream>>>(V2, V2t, RF, Dd);

  // --- low-rank Q/K/V: x @ P (8192x384, K=768), then per-head @ V (K=32) ---
  gemm(xb, Pqt, nullptr, xPq, nullptr, 8192, 384, 768, 768, 384, 0, 0, 0, 0, 0, 1);
  gemm(xb, Pkt, nullptr, xPk, nullptr, 8192, 384, 768, 768, 384, 0, 0, 0, 0, 0, 1);
  gemm(xb, Pvt, nullptr, xPv, nullptr, 8192, 384, 768, 768, 384, 0, 0, 0, 0, 0, 1);
  const long long sVw = (long long)DH * Rr, sCq = BM * DH;
  gemm(xPq, Vqt, nullptr, Qb, bq, 8192, 64, 32, 384, 64, 32, sVw, sCq, 64, 0, Hh);
  gemm(xPk, Vkt, nullptr, Kb, bk, 8192, 64, 32, 384, 64, 32, sVw, sCq, 64, 0, Hh);
  gemm(xPv, Vvt, nullptr, Vb, bv, 8192, 64, 32, 384, 64, 32, sVw, sCq, 64, 0, Hh);
  transpose_kernel<<<512, 256, 0, stream>>>(Vb, Vtb, Mm, DH, (long long)Hh * BM * DH);

  // --- flash attention ---
  attn_kernel<<<dim3(Mm / 64, Bb, Hh), 128, 0, stream>>>(Qb, Kb, Vtb, mask,
                                                         attn, Mm, Dd, 0.125f);

  // --- output projection + residual + LN1 ---
  gemm(attn, Uot, nullptr, attnU, nullptr, 8192, 384, 768, 768, 384, 0, 0, 0, 0, 0, 1);
  gemm(attnU, Vot, y0, nullptr, bo, 8192, 768, 384, 384, 768, 0, 0, 0, 0, 0, 1);
  ln_kernel<<<8192, 256, 0, stream>>>(x, y0, g1, be1, x1f, x1b, Dd);

  // --- low-rank FFN with exact GELU ---
  gemm(x1b, U1t, nullptr, mid, nullptr, 8192, 384, 768, 768, 384, 0, 0, 0, 0, 0, 1);
  gemm(mid, V1t, nullptr, ffn1, b1, 8192, 3072, 384, 384, 3072, 0, 0, 0, 0, 1, 1);
  gemm(ffn1, U2t, nullptr, ffn2, nullptr, 8192, 384, 3072, 3072, 384, 0, 0, 0, 0, 0, 1);
  gemm(ffn2, V2t, y2, nullptr, b2, 8192, 768, 384, 384, 768, 0, 0, 0, 0, 0, 1);

  // --- residual + LN2 -> fp32 output ---
  ln_kernel<<<8192, 256, 0, stream>>>(x1f, y2, g2, be2, (float*)d_out, nullptr, Dd);
}
